// SwinTransformer_4853313044775
// MI455X (gfx1250) — matrix operand order repaired and checked by reading, not compiled, not run on hardware
//
#include <hip/hip_runtime.h>
#include <hip/hip_bf16.h>

typedef __attribute__((ext_vector_type(16))) _Float16 v16h;
typedef __attribute__((ext_vector_type(8)))  float    v8f;

#define DEVINL __device__ __forceinline__

// ---- problem constants -----------------------------------------------------
constexpr int CDIM = 192, NHEAD = 6, HDIM = 32;
constexpr int WSZ = 7, SHFT = 3, IMG = 56, TOKENS = IMG * IMG;   // 3136
constexpr int NTOK = 49, NPAD = 64;           // window tokens, padded to 64
constexpr int TC3 = 576, MLPH = 768;

// LDS row strides (elements), padded to stagger banks
constexpr int XS = 200;   // X / O rows (halves)
constexpr int QS = 584;   // qkv rows (halves)
constexpr int SS = 66;    // S rows (floats)
constexpr int PS = 72;    // P rows (halves)
constexpr int HS = 776;   // MLP hidden rows (halves)

// ---- pre-swizzled weight-fragment workspace layout (halves) ----------------
constexpr int QKVF_TILES = 6 * 36;   // K=192/32, N=576/16
constexpr int PROJF_TILES = 6 * 12;  // 192x192
constexpr int FC1F_TILES = 6 * 48;   // 192x768
constexpr int FC2F_TILES = 24 * 12;  // 768x192
constexpr int QKVF_OFF = 0;
constexpr int PROJF_OFF = QKVF_OFF + QKVF_TILES * 512;
constexpr int FC1F_OFF  = PROJF_OFF + PROJF_TILES * 512;
constexpr int FC2F_OFF  = FC1F_OFF + FC1F_TILES * 512;
constexpr int TOT_TILES = QKVF_TILES + PROJF_TILES + FC1F_TILES + FC2F_TILES; // 864

// ---- WMMA fragment loaders (contiguous k = 16*(lane/16) + s order, wave32) --------------------
// A 16x32 f16: row = lane&15; lanes<16 hold K 0-15, lanes>=16 K 16-31
DEVINL v16h load_a(const _Float16* base, int stride) {
  const int l = threadIdx.x & 31;
  const _Float16* p = base + (l & 15) * stride + ((l < 16) ? 0 : 16);
  v16h a;
#pragma unroll
  for (int j = 0; j < 8; ++j) { a[j] = p[j]; a[8 + j] = p[8 + j]; }
  return a;
}
// B 32x16 f16 from row-major [k][n]: col = lane&15; lanes<16 K 0-15, lanes>=16 K 16-31
DEVINL v16h load_b(const _Float16* base, int stride) {
  const int l = threadIdx.x & 31;
  const _Float16* p = base + (l & 15) + ((l < 16) ? 0 : 16 * stride);
  v16h b;
#pragma unroll
  for (int j = 0; j < 16; ++j) b[j] = p[j * stride];
  return b;
}
// B from transposed source [n][k] (contiguous per lane -> ds_load_b128 pairs)
DEVINL v16h load_bt(const _Float16* base, int stride) {
  const int l = threadIdx.x & 31;
  const _Float16* p = base + (l & 15) * stride + ((l < 16) ? 0 : 16);
  v16h b;
#pragma unroll
  for (int j = 0; j < 16; ++j) b[j] = p[j];
  return b;
}
// B from pre-swizzled global fragment buffer: one 32B contiguous load per lane
DEVINL v16h load_bfrag(const _Float16* buf, int kt, int nt, int NT) {
  return *(const v16h*)(buf + (((kt * NT + nt) * 32) + (threadIdx.x & 31)) * 16);
}

#define WMMA_F16(acc, a, b)                                                    \
  acc = __builtin_amdgcn_wmma_f32_16x16x32_f16(false, a, false, b, (short)0,   \
                                               acc, false, false)

DEVINL float wave_sum(float v) {
#pragma unroll
  for (int off = 16; off; off >>= 1) v += __shfl_xor(v, off, 32);
  return v;
}
DEVINL float wave_max(float v) {
#pragma unroll
  for (int off = 16; off; off >>= 1) v = fmaxf(v, __shfl_xor(v, off, 32));
  return v;
}

// warm L2/L0 with a weight-fragment region (lowers to global_prefetch_b8)
DEVINL void prefetch_region(const _Float16* p, int halves) {
  for (int off = (int)threadIdx.x * 64; off < halves; off += 256 * 64)
    __builtin_prefetch(p + off, 0, 1);
}

// ---- kernel 1: fp32 weights -> f16 pre-swizzled B fragments ----------------
__global__ __launch_bounds__(256) void prep_weights(
    const float* __restrict__ qkv_w, const float* __restrict__ proj_w,
    const float* __restrict__ fc1_w, const float* __restrict__ fc2_w,
    _Float16* __restrict__ wsh) {
  const int g = blockIdx.x * 256 + threadIdx.x;
  const int tile = g >> 5, l = g & 31;
  if (tile >= TOT_TILES) return;
  const float* src; int Ncols, NT, lt; _Float16* dst;
  if (tile < QKVF_TILES) {
    src = qkv_w; Ncols = TC3; NT = 36; lt = tile; dst = wsh + QKVF_OFF;
  } else if (tile < QKVF_TILES + PROJF_TILES) {
    src = proj_w; Ncols = CDIM; NT = 12; lt = tile - QKVF_TILES; dst = wsh + PROJF_OFF;
  } else if (tile < QKVF_TILES + PROJF_TILES + FC1F_TILES) {
    src = fc1_w; Ncols = MLPH; NT = 48; lt = tile - QKVF_TILES - PROJF_TILES; dst = wsh + FC1F_OFF;
  } else {
    src = fc2_w; Ncols = CDIM; NT = 12;
    lt = tile - QKVF_TILES - PROJF_TILES - FC1F_TILES; dst = wsh + FC2F_OFF;
  }
  const int kt = lt / NT, nt = lt % NT;
  const int col = nt * 16 + (l & 15);
  const int kb = kt * 32 + ((l < 16) ? 0 : 16);
  v16h frag;
#pragma unroll
  for (int j = 0; j < 16; ++j) frag[j] = (_Float16)src[(kb + j) * Ncols + col];
  *(v16h*)(dst + (lt * 32 + l) * 16) = frag;
}

// ---- kernel 2: fused LN1 + shifted-window attention + proj + residual ------
__global__ __launch_bounds__(256) void swin_attn(
    const float* __restrict__ x, const float* __restrict__ n1w,
    const float* __restrict__ n1b, const float* __restrict__ qkv_b,
    const float* __restrict__ proj_b, const float* __restrict__ rel_bias,
    const _Float16* __restrict__ wf, float* __restrict__ out) {
  __shared__ _Float16 qkvh[NPAD * QS];
  __shared__ float    Smat[NPAD * SS];
  __shared__ _Float16 Pm[NPAD * PS];
  __shared__ _Float16 XO[NPAD * XS];
  __shared__ unsigned char tokRi[NPAD], tokCi[NPAD], tokSid[NPAD];

  const int bid = blockIdx.x;
  const int batch = bid >> 6;
  const int nw = bid & 63;
  const int wh = nw >> 3, ww = nw & 7;
  const int l = threadIdx.x & 31;
  const int wv = threadIdx.x >> 5;

  prefetch_region(wf + QKVF_OFF, QKVF_TILES * 512);

  if (threadIdx.x < NPAD) {
    const int t = threadIdx.x;
    if (t < NTOK) {
      const int r = t / WSZ, c = t % WSZ;
      tokRi[t] = (unsigned char)r;
      tokCi[t] = (unsigned char)c;
      const int hr = wh * WSZ + r, wr = ww * WSZ + c;
      const int sh = (hr < 49) ? 0 : ((hr < 53) ? 1 : 2);
      const int sw = (wr < 49) ? 0 : ((wr < 53) ? 1 : 2);
      tokSid[t] = (unsigned char)(sh * 3 + sw);
    } else {
      tokRi[t] = 0; tokCi[t] = 0; tokSid[t] = 255;
    }
  }

  // phase 0: LN1 + roll + window gather -> XO (f16), padded rows zeroed
  for (int t = wv; t < NPAD; t += 8) {
    if (t < NTOK) {
      const int r = t / WSZ, c = t % WSZ;
      int h0 = wh * WSZ + r + SHFT; if (h0 >= IMG) h0 -= IMG;
      int w0 = ww * WSZ + c + SHFT; if (w0 >= IMG) w0 -= IMG;
      const float* src = x + (batch * TOKENS + h0 * IMG + w0) * CDIM;
      float v[6]; float s = 0.f;
#pragma unroll
      for (int j = 0; j < 6; ++j) { v[j] = src[j * 32 + l]; s += v[j]; }
      const float mean = wave_sum(s) * (1.f / 192.f);
      float vs = 0.f;
#pragma unroll
      for (int j = 0; j < 6; ++j) { const float d = v[j] - mean; vs += d * d; }
      const float rstd = rsqrtf(wave_sum(vs) * (1.f / 192.f) + 1e-5f);
#pragma unroll
      for (int j = 0; j < 6; ++j) {
        const int idx = j * 32 + l;
        XO[t * XS + idx] = (_Float16)((v[j] - mean) * rstd * n1w[idx] + n1b[idx]);
      }
    } else {
#pragma unroll
      for (int j = 0; j < 6; ++j) XO[t * XS + j * 32 + l] = (_Float16)0.f;
    }
  }
  __syncthreads();

  // phase 1: QKV GEMM, N-register-blocked: B fragments loaded ONCE per column
  for (int nt = wv; nt < 36; nt += 8) {
    v16h bfr[6];
#pragma unroll
    for (int kk = 0; kk < 6; ++kk)
      bfr[kk] = load_bfrag(wf + QKVF_OFF, kk, nt, 36);
    const int cg = nt * 16 + (l & 15);
    const float qb = qkv_b[cg];
    const int rb = (l < 16) ? 0 : 8;
#pragma unroll
    for (int mt = 0; mt < 4; ++mt) {
      v8f acc = {};
#pragma unroll
      for (int kk = 0; kk < 6; ++kk) {
        v16h a = load_a(&XO[mt * 16 * XS + kk * 32], XS);
        WMMA_F16(acc, a, bfr[kk]);
      }
#pragma unroll
      for (int r = 0; r < 8; ++r)
        qkvh[(mt * 16 + rb + r) * QS + cg] = (_Float16)(acc[r] + qb);
    }
  }
  __syncthreads();

  // phase 2: per-head attention
  const float scale = 0.17677669529663687f;  // 1/sqrt(32)
  for (int head = 0; head < NHEAD; ++head) {
    const int colQ = head * HDIM;
    const int colK = CDIM + head * HDIM;
    const int colV = 2 * CDIM + head * HDIM;

    // S = Q K^T (single K=32 WMMA per 16x16 tile) + scale + rel_bias + mask
    for (int tid = wv; tid < 16; tid += 8) {
      const int mt = tid >> 2, nt = tid & 3;
      v16h a = load_a(&qkvh[mt * 16 * QS + colQ], QS);
      v16h b = load_bt(&qkvh[nt * 16 * QS + colK], QS);
      v8f acc = {};
      WMMA_F16(acc, a, b);
      const int colL = l & 15, rb = (l < 16) ? 0 : 8;
      const int ng = nt * 16 + colL;
#pragma unroll
      for (int r = 0; r < 8; ++r) {
        const int mg = mt * 16 + rb + r;
        float val;
        if (mg < NTOK && ng < NTOK) {
          const int idx = ((int)tokRi[mg] - (int)tokRi[ng] + 6) * 13 +
                          ((int)tokCi[mg] - (int)tokCi[ng] + 6);
          const float bias = rel_bias[idx * NHEAD + head];
          const float msk = (tokSid[mg] == tokSid[ng]) ? 0.f : -100.f;
          val = acc[r] * scale + bias + msk;
        } else {
          val = -1e9f;
        }
        Smat[mg * SS + ng] = val;
      }
    }
    __syncthreads();

    // softmax over valid 49 columns; pad rows/cols zeroed
    for (int row = wv; row < NPAD; row += 8) {
      if (row < NTOK) {
        const float v0 = Smat[row * SS + l];
        const float v1 = (l + 32 < NTOK) ? Smat[row * SS + l + 32] : -3e38f;
        const float m = wave_max(fmaxf(v0, v1));
        const float e0 = __expf(v0 - m);
        const float e1 = (l + 32 < NTOK) ? __expf(v1 - m) : 0.f;
        const float inv = 1.f / wave_sum(e0 + e1);
        Pm[row * PS + l] = (_Float16)(e0 * inv);
        Pm[row * PS + l + 32] = (_Float16)(e1 * inv);
      } else {
        Pm[row * PS + l] = (_Float16)0.f;
        Pm[row * PS + l + 32] = (_Float16)0.f;
      }
    }
    __syncthreads();

    // O = P V (8 tiles, one per wave; K=64 -> 2 WMMA steps)
    {
      const int mt = wv >> 1, nt = wv & 1;
      v8f acc = {};
#pragma unroll
      for (int kk = 0; kk < 2; ++kk) {
        v16h a = load_a(&Pm[mt * 16 * PS + kk * 32], PS);
        v16h b = load_b(&qkvh[kk * 32 * QS + colV + nt * 16], QS);
        WMMA_F16(acc, a, b);
      }
      const int colL = l & 15, rb = (l < 16) ? 0 : 8;
#pragma unroll
      for (int r = 0; r < 8; ++r)
        XO[(mt * 16 + rb + r) * XS + head * HDIM + nt * 16 + colL] =
            (_Float16)acc[r];
    }
    __syncthreads();
  }

  // phase 3: proj GEMM (N-blocked, 2 M-tiles per work item) + reverse + roll
  // + residual scatter
  for (int item = wv; item < 24; item += 8) {
    const int nt = item % 12, mh = item / 12;
    v16h bfr[6];
#pragma unroll
    for (int kk = 0; kk < 6; ++kk)
      bfr[kk] = load_bfrag(wf + PROJF_OFF, kk, nt, 12);
    v8f accs[2] = {{}, {}};
#pragma unroll
    for (int kk = 0; kk < 6; ++kk) {
#pragma unroll
      for (int mi = 0; mi < 2; ++mi) {
        v16h a = load_a(&XO[(mh * 2 + mi) * 16 * XS + kk * 32], XS);
        WMMA_F16(accs[mi], a, bfr[kk]);
      }
    }
    const int colL = l & 15, rb = (l < 16) ? 0 : 8;
    const int cg = nt * 16 + colL;
    const float pb = proj_b[cg];
#pragma unroll
    for (int mi = 0; mi < 2; ++mi) {
      const int mt = mh * 2 + mi;
#pragma unroll
      for (int r = 0; r < 8; ++r) {
        const int m = mt * 16 + rb + r;
        if (m < NTOK) {
          int h0 = wh * WSZ + (int)tokRi[m] + SHFT; if (h0 >= IMG) h0 -= IMG;
          int w0 = ww * WSZ + (int)tokCi[m] + SHFT; if (w0 >= IMG) w0 -= IMG;
          const int dst = (batch * TOKENS + h0 * IMG + w0) * CDIM + cg;
          out[dst] = x[dst] + accs[mi][r] + pb;
        }
      }
    }
  }
}

// ---- kernel 3: fused LN2 + MLP (fc1 + GELU + fc2) + residual ---------------
__global__ __launch_bounds__(256) void swin_mlp(
    const float* __restrict__ n2w, const float* __restrict__ n2b,
    const float* __restrict__ fc1_b, const float* __restrict__ fc2_b,
    const _Float16* __restrict__ wf, float* __restrict__ out) {
  __shared__ _Float16 Xh[NPAD * XS];
  __shared__ _Float16 Hd[NPAD * HS];
  const int T0 = blockIdx.x * 64;
  const int l = threadIdx.x & 31, wv = threadIdx.x >> 5;

  prefetch_region(wf + FC1F_OFF, FC1F_TILES * 512);

  // LN2 of x2 (resident in out) -> Xh (f16)
  for (int rr = wv; rr < 64; rr += 8) {
    const float* src = out + (T0 + rr) * CDIM;
    float v[6]; float s = 0.f;
#pragma unroll
    for (int j = 0; j < 6; ++j) { v[j] = src[j * 32 + l]; s += v[j]; }
    const float mean = wave_sum(s) * (1.f / 192.f);
    float vs = 0.f;
#pragma unroll
    for (int j = 0; j < 6; ++j) { const float d = v[j] - mean; vs += d * d; }
    const float rstd = rsqrtf(wave_sum(vs) * (1.f / 192.f) + 1e-5f);
#pragma unroll
    for (int j = 0; j < 6; ++j) {
      const int idx = j * 32 + l;
      Xh[rr * XS + idx] = (_Float16)((v[j] - mean) * rstd * n2w[idx] + n2b[idx]);
    }
  }
  __syncthreads();

  // fc1 (N-blocked) + bias + exact GELU -> Hd (f16)
  for (int nt = wv; nt < 48; nt += 8) {
    v16h bfr[6];
#pragma unroll
    for (int kk = 0; kk < 6; ++kk)
      bfr[kk] = load_bfrag(wf + FC1F_OFF, kk, nt, 48);
    const int cg = nt * 16 + (l & 15);
    const float b1 = fc1_b[cg];
    const int rb = (l < 16) ? 0 : 8;
#pragma unroll
    for (int mt = 0; mt < 4; ++mt) {
      v8f acc = {};
#pragma unroll
      for (int kk = 0; kk < 6; ++kk) {
        v16h a = load_a(&Xh[mt * 16 * XS + kk * 32], XS);
        WMMA_F16(acc, a, bfr[kk]);
      }
#pragma unroll
      for (int r = 0; r < 8; ++r) {
        const float v = acc[r] + b1;
        const float g = 0.5f * v * (1.f + erff(v * 0.70710678118654752f));
        Hd[(mt * 16 + rb + r) * HS + cg] = (_Float16)g;
      }
    }
  }
  __syncthreads();

  // fc2: K=768 in 4 chunks of 6; B chunk shared across 2 M-tiles per item
  for (int item = wv; item < 24; item += 8) {
    const int nt = item % 12, mh = item / 12;
    v8f accs[2] = {{}, {}};
#pragma unroll
    for (int kc = 0; kc < 4; ++kc) {
      v16h bfr[6];
#pragma unroll
      for (int j = 0; j < 6; ++j)
        bfr[j] = load_bfrag(wf + FC2F_OFF, kc * 6 + j, nt, 12);
#pragma unroll
      for (int j = 0; j < 6; ++j) {
#pragma unroll
        for (int mi = 0; mi < 2; ++mi) {
          v16h a = load_a(&Hd[(mh * 2 + mi) * 16 * HS + (kc * 6 + j) * 32], HS);
          WMMA_F16(accs[mi], a, bfr[j]);
        }
      }
    }
    const int cg = nt * 16 + (l & 15);
    const float b2 = fc2_b[cg];
    const int rb = (l < 16) ? 0 : 8;
#pragma unroll
    for (int mi = 0; mi < 2; ++mi) {
      const int mt = mh * 2 + mi;
#pragma unroll
      for (int r = 0; r < 8; ++r) {
        const int m = mt * 16 + rb + r;
        const int idx = (T0 + m) * CDIM + cg;
        out[idx] = out[idx] + accs[mi][r] + b2;
      }
    }
  }
}

// ---- host launch -----------------------------------------------------------
extern "C" void kernel_launch(void* const* d_in, const int* in_sizes, int n_in,
                              void* d_out, int out_size, void* d_ws,
                              size_t ws_size, hipStream_t stream) {
  const float* x       = (const float*)d_in[0];
  const float* norm1_w = (const float*)d_in[1];
  const float* norm1_b = (const float*)d_in[2];
  const float* qkv_w   = (const float*)d_in[3];
  const float* qkv_b   = (const float*)d_in[4];
  const float* proj_w  = (const float*)d_in[5];
  const float* proj_b  = (const float*)d_in[6];
  const float* relb    = (const float*)d_in[7];
  const float* norm2_w = (const float*)d_in[8];
  const float* norm2_b = (const float*)d_in[9];
  const float* fc1_w   = (const float*)d_in[10];
  const float* fc1_b   = (const float*)d_in[11];
  const float* fc2_w   = (const float*)d_in[12];
  const float* fc2_b   = (const float*)d_in[13];
  float* out = (float*)d_out;
  _Float16* wsh = (_Float16*)d_ws;

  prep_weights<<<(TOT_TILES * 32 + 255) / 256, 256, 0, stream>>>(
      qkv_w, proj_w, fc1_w, fc2_w, wsh);
  swin_attn<<<64 * 64, 256, 0, stream>>>(x, norm1_w, norm1_b, qkv_b, proj_b,
                                         relb, wsh, out);
  swin_mlp<<<TOKENS * 64 / 64, 256, 0, stream>>>(norm2_w, norm2_b, fc1_b,
                                                 fc2_b, wsh, out);
}
